// CrossAttentionLayer_54546084659647
// MI455X (gfx1250) — matrix operand order repaired and checked by reading, not compiled, not run on hardware
//
#include <hip/hip_runtime.h>

typedef __bf16 bf16;
typedef __attribute__((ext_vector_type(16))) __bf16 v16bf;
typedef __attribute__((ext_vector_type(8)))  float  v8f;
typedef __attribute__((ext_vector_type(4)))  int    v4i;

union FragB16 {
    v16bf v;
    uint4 u[2];
};

#define D_DIM   1024
#define N_ROWS  65536
#define BKH     72      // 64 K-halfwords + 8 pad (row stride 144 B, 16B-aligned)

#if defined(__HIP_DEVICE_COMPILE__) && __has_builtin(__builtin_amdgcn_global_load_async_to_lds_b128)
#define USE_ASYNC_LDS 1
#else
#define USE_ASYNC_LDS 0
#endif

// Global (AS1) / LDS (AS3) typed pointer casts for the async-copy builtin.
#define GAS(p) ((__attribute__((address_space(1))) v4i*)(p))
#define LAS(p) ((__attribute__((address_space(3))) v4i*)(p))

__device__ __forceinline__ void wait_async_all()
{
#if USE_ASYNC_LDS
#if __has_builtin(__builtin_amdgcn_s_wait_asynccnt)
    __builtin_amdgcn_s_wait_asynccnt(0);
#else
    asm volatile("s_wait_asynccnt 0x0" ::: "memory");
#endif
#endif
}

// Stage one 128x64 bf16 A tile into LDS (async global->LDS when available).
__device__ __forceinline__ void stage_tile(const bf16* __restrict__ Ag,
                                           bf16 (&As)[128][BKH],
                                           size_t m0, int k0, int tid)
{
    #pragma unroll
    for (int p = 0; p < 4; ++p) {
        int q   = tid + p * 256;
        int row = q >> 3;          // 0..127
        int col = (q & 7) * 8;     // 0..56, 8 bf16 = 16 B
        const bf16* g = Ag + (m0 + row) * D_DIM + k0 + col;
        bf16* l = &As[row][col];
#if USE_ASYNC_LDS
        __builtin_amdgcn_global_load_async_to_lds_b128(GAS(g), LAS(l), 0, 0);
#else
        *(uint4*)l = *(const uint4*)g;
#endif
    }
}

// Shared 128x128 (per WG) x K=1024 bf16 WMMA mainloop.
// Wave tile 64x32 = 4x2 accumulators of v_wmma_f32_16x16x32_bf16.
// B fragments come straight from pre-shuffled (fragment-layout) global weights.
__device__ __forceinline__ void gemm_mainloop(const bf16* __restrict__ Ag,
                                              const bf16* __restrict__ WFz,
                                              bf16 (&As)[2][128][BKH],
                                              size_t m0, int n0, int tid,
                                              v8f (&acc)[4][2])
{
    const int wid = tid >> 5, lane = tid & 31;
    const int wm = wid >> 2, wn = wid & 3;
    const int lane16 = lane & 15;
    const int kbase  = (lane < 16) ? 0 : 16;

    stage_tile(Ag, As[0], m0, 0, tid);
    wait_async_all();
    __syncthreads();

    for (int kt = 0; kt < 16; ++kt) {
        const int cur = kt & 1;
        if (kt + 1 < 16)
            stage_tile(Ag, As[cur ^ 1], m0, (kt + 1) * 64, tid);

        #pragma unroll
        for (int kb = 0; kb < 2; ++kb) {
            const int ktile = kt * 2 + kb;            // 32-wide K block idx
            FragB16 bfr[2];
            #pragma unroll
            for (int j = 0; j < 2; ++j) {
                int nt = (n0 >> 4) + wn * 2 + j;
                const bf16* bp =
                    WFz + (size_t)(ktile * 64 + nt) * 512 + lane * 16;
                bfr[j].u[0] = *(const uint4*)(bp);
                bfr[j].u[1] = *(const uint4*)(bp + 8);
            }
            #pragma unroll
            for (int i = 0; i < 4; ++i) {
                FragB16 afr;
                const bf16* ap =
                    &As[cur][wm * 64 + i * 16 + lane16][kb * 32 + kbase];
                afr.u[0] = *(const uint4*)(ap);
                afr.u[1] = *(const uint4*)(ap + 8);
                #pragma unroll
                for (int j = 0; j < 2; ++j) {
                    acc[i][j] = __builtin_amdgcn_wmma_f32_16x16x32_bf16(
                        false, afr.v, false, bfr[j].v, (short)0, acc[i][j],
                        false, false);
                }
            }
        }
        wait_async_all();
        __syncthreads();
    }
}

// ---------------------------------------------------------------------------
// Kernel: cast f32 inputs to bf16 (query -> Qc, key_value -> KVc).
// ---------------------------------------------------------------------------
__global__ __launch_bounds__(256) void cast_inputs(
    const float* __restrict__ q, const float* __restrict__ kv,
    bf16* __restrict__ Qc, bf16* __restrict__ KVc)
{
    const float* src = blockIdx.z ? kv : q;
    bf16* dst = blockIdx.z ? KVc : Qc;
    size_t i = ((size_t)blockIdx.x * 256 + threadIdx.x) * 8;
    float4 a = *(const float4*)(src + i);
    float4 b = *(const float4*)(src + i + 4);
    union { bf16 h[8]; uint4 u; } o;
    o.h[0] = (bf16)a.x; o.h[1] = (bf16)a.y; o.h[2] = (bf16)a.z; o.h[3] = (bf16)a.w;
    o.h[4] = (bf16)b.x; o.h[5] = (bf16)b.y; o.h[6] = (bf16)b.z; o.h[7] = (bf16)b.w;
    *(uint4*)(dst + i) = o.u;
}

// ---------------------------------------------------------------------------
// Kernel: cast 4 weight matrices f32 -> bf16 in WMMA B-fragment layout.
//   block (kt in [0,32), nt in [0,64)): lane = ((k&16)?16:0)+(n&15), elem = k&15
// ---------------------------------------------------------------------------
__global__ __launch_bounds__(256) void prep_weights(
    const float* __restrict__ Wq, const float* __restrict__ Wk,
    const float* __restrict__ Wv, const float* __restrict__ Wo,
    bf16* __restrict__ WF)
{
    size_t idx = (size_t)blockIdx.x * 256 + threadIdx.x;   // 4 * 1M total
    int m = (int)(idx >> 20);
    int k = (int)(idx >> 10) & 1023;
    int n = (int)idx & 1023;
    const float* W = (m == 0) ? Wq : (m == 1) ? Wk : (m == 2) ? Wv : Wo;
    float val = W[(size_t)k * D_DIM + n];
    int kt = k >> 5, kin = k & 31;
    int nt = n >> 4, nin = n & 15;
    int lane = ((kin & 16) ? 16 : 0) + nin;
    int elem = kin & 15;
    size_t dst = ((size_t)m << 20) + (size_t)(kt * 64 + nt) * 512 + lane * 16 + elem;
    WF[dst] = (bf16)val;
}

// ---------------------------------------------------------------------------
// Kernel: fused Q/K/V projection GEMM, grid (8, 512, 3).
// ---------------------------------------------------------------------------
__global__ __launch_bounds__(256) void gemm_qkv(
    const bf16* __restrict__ Qc, const bf16* __restrict__ KVc,
    const bf16* __restrict__ WF,
    const float* __restrict__ bq, const float* __restrict__ bk,
    const float* __restrict__ bv,
    bf16* __restrict__ Qb, bf16* __restrict__ Kb, bf16* __restrict__ Vb)
{
    const int z = blockIdx.z;
    const bf16*  A    = (z == 0) ? Qc : KVc;
    const float* bias = (z == 0) ? bq : (z == 1) ? bk : bv;
    const bf16*  WFz  = WF + ((size_t)z << 20);
    bf16*        Out  = (z == 0) ? Qb : (z == 1) ? Kb : Vb;

    __shared__ bf16 As[2][128][BKH];

    const int n0 = blockIdx.x * 128;
    const size_t m0 = (size_t)blockIdx.y * 128;
    const int tid = threadIdx.x;
    const int wid = tid >> 5, lane = tid & 31;
    const int wm = wid >> 2, wn = wid & 3;
    const int lane16 = lane & 15;

    v8f acc[4][2] = {};
    gemm_mainloop(A, WFz, As, m0, n0, tid, acc);

    #pragma unroll
    for (int i = 0; i < 4; ++i) {
        #pragma unroll
        for (int j = 0; j < 2; ++j) {
            int col = n0 + wn * 32 + j * 16 + lane16;
            float bval = bias[col];
            #pragma unroll
            for (int r = 0; r < 8; ++r) {
                size_t row = m0 + wm * 64 + i * 16 + r + ((lane < 16) ? 0 : 8);
                Out[row * D_DIM + col] = (bf16)(acc[i][j][r] + bval);
            }
        }
    }
}

// ---------------------------------------------------------------------------
// Kernel: per-row 8x8 head attention; one wave per row.
// ---------------------------------------------------------------------------
__global__ __launch_bounds__(256) void attn_kernel(
    const bf16* __restrict__ Qb, const bf16* __restrict__ Kb,
    const bf16* __restrict__ Vb, bf16* __restrict__ Ab)
{
    const int wid = threadIdx.x >> 5, lane = threadIdx.x & 31;
    const size_t row = (size_t)blockIdx.x * 8 + wid;
    const bf16* q = Qb + row * D_DIM;
    const bf16* k = Kb + row * D_DIM;
    const bf16* v = Vb + row * D_DIM;

    float qv[8][4], kf[8][4], vf[8][4];
    #pragma unroll
    for (int h = 0; h < 8; ++h) {
        union { uint2 u; bf16 h4[4]; } a, b, c;
        a.u = *(const uint2*)(q + h * 128 + lane * 4);
        b.u = *(const uint2*)(k + h * 128 + lane * 4);
        c.u = *(const uint2*)(v + h * 128 + lane * 4);
        #pragma unroll
        for (int i = 0; i < 4; ++i) {
            qv[h][i] = (float)a.h4[i];
            kf[h][i] = (float)b.h4[i];
            vf[h][i] = (float)c.h4[i];
        }
    }

    #pragma unroll
    for (int h = 0; h < 8; ++h) {
        float s[8];
        #pragma unroll
        for (int g = 0; g < 8; ++g) {
            float p = 0.f;
            #pragma unroll
            for (int i = 0; i < 4; ++i) p += qv[h][i] * kf[g][i];
            #pragma unroll
            for (int off = 16; off > 0; off >>= 1)
                p += __shfl_xor(p, off, 32);
            s[g] = p * 0.08838834764831845f;  // 1/sqrt(128)
        }
        float mx = s[0];
        #pragma unroll
        for (int g = 1; g < 8; ++g) mx = fmaxf(mx, s[g]);
        float e[8], sum = 0.f;
        #pragma unroll
        for (int g = 0; g < 8; ++g) { e[g] = __expf(s[g] - mx); sum += e[g]; }
        float inv = 1.0f / sum;
        float o[4] = {0.f, 0.f, 0.f, 0.f};
        #pragma unroll
        for (int g = 0; g < 8; ++g) {
            #pragma unroll
            for (int i = 0; i < 4; ++i) o[i] += e[g] * vf[g][i];
        }
        union { uint2 u; bf16 h4[4]; } st;
        #pragma unroll
        for (int i = 0; i < 4; ++i) st.h4[i] = (bf16)(o[i] * inv);
        *(uint2*)(Ab + row * D_DIM + h * 128 + lane * 4) = st.u;
    }
}

// ---------------------------------------------------------------------------
// Kernel: output GEMM: attended @ Wo + bo + residual(query f32) -> f32 d_out.
// ---------------------------------------------------------------------------
__global__ __launch_bounds__(256) void gemm_o(
    const bf16* __restrict__ Ab, const bf16* __restrict__ WoF,
    const float* __restrict__ bo, const float* __restrict__ query,
    float* __restrict__ out)
{
    __shared__ bf16 As[2][128][BKH];

    const int n0 = blockIdx.x * 128;
    const size_t m0 = (size_t)blockIdx.y * 128;
    const int tid = threadIdx.x;
    const int wid = tid >> 5, lane = tid & 31;
    const int wm = wid >> 2, wn = wid & 3;
    const int lane16 = lane & 15;

    v8f acc[4][2] = {};
    gemm_mainloop(Ab, WoF, As, m0, n0, tid, acc);

    #pragma unroll
    for (int i = 0; i < 4; ++i) {
        #pragma unroll
        for (int j = 0; j < 2; ++j) {
            int col = n0 + wn * 32 + j * 16 + lane16;
            float bval = bo[col];
            #pragma unroll
            for (int r = 0; r < 8; ++r) {
                size_t row = m0 + wm * 64 + i * 16 + r + ((lane < 16) ? 0 : 8);
                out[row * D_DIM + col] =
                    acc[i][j][r] + bval + query[row * D_DIM + col];
            }
        }
    }
}

// ---------------------------------------------------------------------------
// Kernel: in-place LayerNorm over last dim (1024); one wave per row.
// ---------------------------------------------------------------------------
__global__ __launch_bounds__(256) void ln_kernel(
    float* __restrict__ y, const float* __restrict__ gamma,
    const float* __restrict__ beta)
{
    const int wid = threadIdx.x >> 5, lane = threadIdx.x & 31;
    const size_t row = (size_t)blockIdx.x * 8 + wid;
    float* p = y + row * D_DIM;

    float x[32];
    float s = 0.f;
    #pragma unroll
    for (int c = 0; c < 8; ++c) {
        float4 f = *(const float4*)(p + c * 128 + lane * 4);
        x[c * 4 + 0] = f.x; x[c * 4 + 1] = f.y;
        x[c * 4 + 2] = f.z; x[c * 4 + 3] = f.w;
        s += f.x + f.y + f.z + f.w;
    }
    #pragma unroll
    for (int off = 16; off > 0; off >>= 1) s += __shfl_xor(s, off, 32);
    const float mu = s * (1.0f / 1024.0f);

    float vs = 0.f;
    #pragma unroll
    for (int i = 0; i < 32; ++i) { float d = x[i] - mu; vs += d * d; }
    #pragma unroll
    for (int off = 16; off > 0; off >>= 1) vs += __shfl_xor(vs, off, 32);
    const float rstd = rsqrtf(vs * (1.0f / 1024.0f) + 1e-5f);

    #pragma unroll
    for (int c = 0; c < 8; ++c) {
        float4 g = *(const float4*)(gamma + c * 128 + lane * 4);
        float4 b = *(const float4*)(beta  + c * 128 + lane * 4);
        float4 o;
        o.x = (x[c * 4 + 0] - mu) * rstd * g.x + b.x;
        o.y = (x[c * 4 + 1] - mu) * rstd * g.y + b.y;
        o.z = (x[c * 4 + 2] - mu) * rstd * g.z + b.z;
        o.w = (x[c * 4 + 3] - mu) * rstd * g.w + b.w;
        *(float4*)(p + c * 128 + lane * 4) = o;
    }
}

// ---------------------------------------------------------------------------
extern "C" void kernel_launch(void* const* d_in, const int* in_sizes, int n_in,
                              void* d_out, int out_size, void* d_ws,
                              size_t ws_size, hipStream_t stream)
{
    (void)in_sizes; (void)n_in; (void)out_size; (void)ws_size;
    const float* query = (const float*)d_in[0];
    const float* keyv  = (const float*)d_in[1];
    const float* Wq    = (const float*)d_in[2];
    const float* bq    = (const float*)d_in[3];
    const float* Wk    = (const float*)d_in[4];
    const float* bk    = (const float*)d_in[5];
    const float* Wv    = (const float*)d_in[6];
    const float* bv    = (const float*)d_in[7];
    const float* Wo    = (const float*)d_in[8];
    const float* bo    = (const float*)d_in[9];
    const float* gamma = (const float*)d_in[10];
    const float* beta  = (const float*)d_in[11];
    float* out = (float*)d_out;

    char* ws = (char*)d_ws;
    bf16* WF  = (bf16*)ws;                        //   8 MB, 4 x 1M bf16 (frag layout)
    bf16* Qc  = (bf16*)(ws + (8ull   << 20));     // 128 MB bf16 query cast
    bf16* KVc = (bf16*)(ws + (136ull << 20));     // 128 MB bf16 key_value cast
    bf16* Qb  = (bf16*)(ws + (264ull << 20));     // 128 MB
    bf16* Kb  = (bf16*)(ws + (392ull << 20));     // 128 MB
    bf16* Vb  = (bf16*)(ws + (520ull << 20));     // 128 MB
    bf16* Ab  = Qc;   // reuse: query cast dead after gemm_qkv

    cast_inputs<<<dim3(32768, 1, 2), 256, 0, stream>>>(query, keyv, Qc, KVc);
    prep_weights<<<16384, 256, 0, stream>>>(Wq, Wk, Wv, Wo, WF);
    gemm_qkv<<<dim3(8, 512, 3), 256, 0, stream>>>(Qc, KVc, WF,
                                                  bq, bk, bv, Qb, Kb, Vb);
    attn_kernel<<<8192, 256, 0, stream>>>(Qb, Kb, Vb, Ab);
    gemm_o<<<dim3(8, 512, 1), 256, 0, stream>>>(Ab, WF + (3ull << 20),
                                                bo, query, out);
    ln_kernel<<<8192, 256, 0, stream>>>(out, gamma, beta);
}
